// TreeGPTBlock_10282151707528
// MI455X (gfx1250) — matrix operand order repaired and checked by reading, not compiled, not run on hardware
//
#include <hip/hip_runtime.h>

// ---------------------------------------------------------------------------
// TreeGPT block for MI455X (gfx1250): f16-in / f32-acc WMMA everywhere.
// B=8, L=2048, D=512, H=8, dh=64, TREE_ITERS=2, M = B*L = 16384.
// GEMM K-loops are software-pipelined: next iteration's A/B fragments are
// issued before the current 8 WMMAs so loads overlap the XDL pipe.
// ---------------------------------------------------------------------------

typedef __attribute__((ext_vector_type(16))) _Float16 v16h;
typedef __attribute__((ext_vector_type(8)))  _Float16 v8h;
typedef __attribute__((ext_vector_type(8)))  float    v8f;

#define WMMA_F32_F16(A, Bf, C) \
  __builtin_amdgcn_wmma_f32_16x16x32_f16(false, (A), false, (Bf), (short)0, (C), false, false)

// A-fragment (16x32 f16, row-major source, row = lane&15):
//   lanes 0-15 : K = {k0+0..7, k0+16..23};  lanes 16-31 : K = {k0+8..15, k0+24..31}
// caller passes p = rowbase + k0 + (lane>>4)*8
__device__ __forceinline__ v16h ldA(const _Float16* p) {
  union { v16h v; v8h h[2]; } u;
  u.h[0] = *(const v8h*)(p);
  u.h[1] = *(const v8h*)(p + 16);
  return u.v;
}
// B-fragment (32x16 f16): lane&15 = column n, same K split as A;
//   lanes 0-15: K = {k0+0..7, k0+16..23} ; lanes 16-31: K = {k0+8..15, k0+24..31}
// caller passes p = colbase(n row of W^T) + k0 + (lane>>4)*8
__device__ __forceinline__ v16h ldB(const _Float16* p) {
  union { v16h v; v8h h[2]; } u;
  u.h[0] = *(const v8h*)(p);
  u.h[1] = *(const v8h*)(p + 16);
  return u.v;
}

// ---------------------------------------------------------------------------
// Software-pipelined 32(M) x 64(N) x K wave tile:
//  a0/a1: A row bases (already + hi*8), b0: W^T col base (already + hi*8),
//  ldb: W^T row stride. Accumulates into acc[2][4].
// ---------------------------------------------------------------------------
__device__ __forceinline__ void gemm_tile(const _Float16* __restrict__ a0,
                                          const _Float16* __restrict__ a1,
                                          const _Float16* __restrict__ b0,
                                          int K, int ldb, v8f (&acc)[2][4]) {
  v16h Af0 = ldA(a0);
  v16h Af1 = ldA(a1);
  v16h Bf0 = ldB(b0 + 0 * 16 * ldb);
  v16h Bf1 = ldB(b0 + 1 * 16 * ldb);
  v16h Bf2 = ldB(b0 + 2 * 16 * ldb);
  v16h Bf3 = ldB(b0 + 3 * 16 * ldb);
  for (int k = 32; k < K; k += 32) {
    // issue next iteration's 12 b128 loads before this iteration's 8 WMMAs
    v16h nA0 = ldA(a0 + k);
    v16h nA1 = ldA(a1 + k);
    v16h nB0 = ldB(b0 + 0 * 16 * ldb + k);
    v16h nB1 = ldB(b0 + 1 * 16 * ldb + k);
    v16h nB2 = ldB(b0 + 2 * 16 * ldb + k);
    v16h nB3 = ldB(b0 + 3 * 16 * ldb + k);
    acc[0][0] = WMMA_F32_F16(Af0, Bf0, acc[0][0]);
    acc[1][0] = WMMA_F32_F16(Af1, Bf0, acc[1][0]);
    acc[0][1] = WMMA_F32_F16(Af0, Bf1, acc[0][1]);
    acc[1][1] = WMMA_F32_F16(Af1, Bf1, acc[1][1]);
    acc[0][2] = WMMA_F32_F16(Af0, Bf2, acc[0][2]);
    acc[1][2] = WMMA_F32_F16(Af1, Bf2, acc[1][2]);
    acc[0][3] = WMMA_F32_F16(Af0, Bf3, acc[0][3]);
    acc[1][3] = WMMA_F32_F16(Af1, Bf3, acc[1][3]);
    Af0 = nA0; Af1 = nA1; Bf0 = nB0; Bf1 = nB1; Bf2 = nB2; Bf3 = nB3;
  }
  acc[0][0] = WMMA_F32_F16(Af0, Bf0, acc[0][0]);
  acc[1][0] = WMMA_F32_F16(Af1, Bf0, acc[1][0]);
  acc[0][1] = WMMA_F32_F16(Af0, Bf1, acc[0][1]);
  acc[1][1] = WMMA_F32_F16(Af1, Bf1, acc[1][1]);
  acc[0][2] = WMMA_F32_F16(Af0, Bf2, acc[0][2]);
  acc[1][2] = WMMA_F32_F16(Af1, Bf2, acc[1][2]);
  acc[0][3] = WMMA_F32_F16(Af0, Bf3, acc[0][3]);
  acc[1][3] = WMMA_F32_F16(Af1, Bf3, acc[1][3]);
}

// ---------------------------------------------------------------------------
// Weight transpose + f32->f16:  WT[n*K + k] = W[k*N + n]
// ---------------------------------------------------------------------------
__global__ void k_transpose(const float* __restrict__ W, _Float16* __restrict__ WT,
                            int K, int N) {
  int idx = blockIdx.x * 256 + threadIdx.x;
  if (idx >= K * N) return;
  int k = idx / N, n = idx - k * N;
  WT[n * K + k] = (_Float16)W[k * N + n];
}

// ---------------------------------------------------------------------------
// LayerNorm over D=512: one wave per row, 16 elems/lane, shfl reductions.
// ---------------------------------------------------------------------------
__global__ __launch_bounds__(256) void k_layernorm(
    const float* __restrict__ in, const float* __restrict__ w,
    const float* __restrict__ bvec, float* __restrict__ out32,
    _Float16* __restrict__ out16) {
  int lane = threadIdx.x & 31;
  int wave = threadIdx.x >> 5;
  int row  = blockIdx.x * 8 + wave;
  const float* p = in + (long)row * 512;
  float vals[16];
  float s = 0.f, s2 = 0.f;
#pragma unroll
  for (int i = 0; i < 16; ++i) {
    float v = p[lane + i * 32];
    vals[i] = v; s += v; s2 += v * v;
  }
#pragma unroll
  for (int off = 16; off >= 1; off >>= 1) {
    s  += __shfl_xor(s,  off);
    s2 += __shfl_xor(s2, off);
  }
  float mean = s * (1.f / 512.f);
  float var  = s2 * (1.f / 512.f) - mean * mean;
  float rstd = rsqrtf(var + 1e-5f);
#pragma unroll
  for (int i = 0; i < 16; ++i) {
    int c = lane + i * 32;
    float v = (vals[i] - mean) * rstd * w[c] + bvec[c];
    if (out32) out32[(long)row * 512 + c] = v;
    if (out16) out16[(long)row * 512 + c] = (_Float16)v;
  }
}

// ---------------------------------------------------------------------------
// QKV GEMM: [16384,512] x WqkvT[1536,512] -> scatter into Qh/Kh (head-major,
// [BH, L, 64]) and VT (head-major transposed, [BH, 64, L]). Bias fused.
// Block 128 = 4 waves, wave tile 32x64, block tile 64x128.
// ---------------------------------------------------------------------------
__global__ __launch_bounds__(128) void k_gemm_qkv(
    const _Float16* __restrict__ A, const _Float16* __restrict__ Bt,
    const float* __restrict__ bias, _Float16* __restrict__ Qh,
    _Float16* __restrict__ Kh, _Float16* __restrict__ VT) {
  const int K = 512;
  int lane = threadIdx.x & 31, wave = threadIdx.x >> 5;
  int mw = wave >> 1, nw = wave & 1;
  int m0 = blockIdx.x * 64 + mw * 32;
  int n0 = blockIdx.y * 128 + nw * 64;
  int r15 = lane & 15, hi = lane >> 4;
  v8f acc[2][4] = {};
  gemm_tile(A + (m0 + r15) * K + hi * 8, A + (m0 + 16 + r15) * K + hi * 8,
            Bt + (n0 + r15) * K + hi * 8, K, K, acc);
#pragma unroll
  for (int i = 0; i < 2; ++i)
#pragma unroll
    for (int j = 0; j < 4; ++j)
#pragma unroll
      for (int r = 0; r < 8; ++r) {
        int m = m0 + i * 16 + hi * 8 + r;
        int n = n0 + j * 16 + r15;
        float v = acc[i][j][r] + bias[n];
        int b = m >> 11, l = m & 2047;
        int sec = n >> 9, cc = n & 511;
        int hh = cc >> 6, d = cc & 63;
        int bh = b * 8 + hh;
        if (sec == 0)      Qh[(bh * 2048 + l) * 64 + d] = (_Float16)v;
        else if (sec == 1) Kh[(bh * 2048 + l) * 64 + d] = (_Float16)v;
        else               VT[(bh * 64 + d) * 2048 + l] = (_Float16)v;
      }
}

// ---------------------------------------------------------------------------
// Flash attention (causal). Grid: (B*H, L/64), 4 waves/WG, 16 q-rows/wave.
// Scores via WMMA (two 16-key tiles per 32-key block), online softmax with
// width-16 shfl row reductions, P through per-wave LDS tile, PV via WMMA.
// ---------------------------------------------------------------------------
__global__ __launch_bounds__(128) void k_attn(
    const _Float16* __restrict__ Qh, const _Float16* __restrict__ Kh,
    const _Float16* __restrict__ VT, _Float16* __restrict__ O) {
  __shared__ _Float16 lds[4][16 * 32];
  int lane = threadIdx.x & 31, wave = threadIdx.x >> 5;
  int bh = blockIdx.x;
  int b = bh >> 3, hh = bh & 7;
  int q0 = blockIdx.y * 64 + wave * 16;
  int r15 = lane & 15, hi = lane >> 4;

  const _Float16* Qb = Qh + (long)bh * 2048 * 64;
  const _Float16* Kb = Kh + (long)bh * 2048 * 64;
  const _Float16* Vb = VT + (long)bh * 64 * 2048;

  const _Float16* qrow = Qb + (q0 + r15) * 64;
  v16h Qf0 = ldA(qrow + 0  + hi * 8);
  v16h Qf1 = ldA(qrow + 32 + hi * 8);

  float mi[8], li[8];
  v8f Oc[4] = {};
#pragma unroll
  for (int r = 0; r < 8; ++r) { mi[r] = -1e30f; li[r] = 0.f; }

  _Float16* pbuf = lds[wave];

  for (int kb = 0; kb < q0 + 16; kb += 32) {
    v8f S[2] = {};
#pragma unroll
    for (int t = 0; t < 2; ++t) {
      const _Float16* krow = Kb + (kb + t * 16 + r15) * 64;
      v16h B0 = ldB(krow + hi * 8);        // d 0..31
      v16h B1 = ldB(krow + 32 + hi * 8);   // d 32..63
      S[t] = WMMA_F32_F16(Qf0, B0, S[t]);
      S[t] = WMMA_F32_F16(Qf1, B1, S[t]);
    }
#pragma unroll
    for (int r = 0; r < 8; ++r) {
      int q_abs = q0 + hi * 8 + r;
      int n0 = kb + r15, n1 = kb + 16 + r15;
      float s0 = (n0 <= q_abs) ? S[0][r] * 0.125f : -1e30f;
      float s1 = (n1 <= q_abs) ? S[1][r] * 0.125f : -1e30f;
      float mx = fmaxf(s0, s1);
#pragma unroll
      for (int off = 8; off >= 1; off >>= 1)
        mx = fmaxf(mx, __shfl_xor(mx, off, 16));
      float mnew = fmaxf(mi[r], mx);
      float corr = __expf(mi[r] - mnew);
      float p0 = __expf(s0 - mnew);
      float p1 = __expf(s1 - mnew);
      float rs = p0 + p1;
#pragma unroll
      for (int off = 8; off >= 1; off >>= 1)
        rs += __shfl_xor(rs, off, 16);
      li[r] = li[r] * corr + rs;
      mi[r] = mnew;
#pragma unroll
      for (int c = 0; c < 4; ++c) Oc[c][r] *= corr;
      int row = hi * 8 + r;
      pbuf[row * 32 + r15]      = (_Float16)p0;
      pbuf[row * 32 + 16 + r15] = (_Float16)p1;
    }
    v16h Pf = ldA(pbuf + r15 * 32 + hi * 8);
#pragma unroll
    for (int c = 0; c < 4; ++c) {
      const _Float16* vrow = Vb + (c * 16 + r15) * 2048 + kb;
      v16h Bf = ldB(vrow + hi * 8);
      Oc[c] = WMMA_F32_F16(Pf, Bf, Oc[c]);
    }
  }
#pragma unroll
  for (int c = 0; c < 4; ++c)
#pragma unroll
    for (int r = 0; r < 8; ++r) {
      int m = q0 + hi * 8 + r;
      float v = Oc[c][r] / li[r];
      O[((long)b * 2048 + m) * 512 + hh * 64 + c * 16 + r15] = (_Float16)v;
    }
}

// ---------------------------------------------------------------------------
// h = x + o @ Wo^T + bo   (f32 out)
// ---------------------------------------------------------------------------
__global__ __launch_bounds__(128) void k_gemm_out(
    const _Float16* __restrict__ A, const _Float16* __restrict__ Bt,
    const float* __restrict__ bias, const float* __restrict__ x,
    float* __restrict__ hout) {
  const int K = 512;
  int lane = threadIdx.x & 31, wave = threadIdx.x >> 5;
  int mw = wave >> 1, nw = wave & 1;
  int m0 = blockIdx.x * 64 + mw * 32;
  int n0 = blockIdx.y * 128 + nw * 64;
  int r15 = lane & 15, hi = lane >> 4;
  v8f acc[2][4] = {};
  gemm_tile(A + (m0 + r15) * K + hi * 8, A + (m0 + 16 + r15) * K + hi * 8,
            Bt + (n0 + r15) * K + hi * 8, K, K, acc);
#pragma unroll
  for (int i = 0; i < 2; ++i)
#pragma unroll
    for (int j = 0; j < 4; ++j)
#pragma unroll
      for (int r = 0; r < 8; ++r) {
        int m = m0 + i * 16 + hi * 8 + r;
        int n = n0 + j * 16 + r15;
        long idx = (long)m * 512 + n;
        hout[idx] = x[idx] + acc[i][j][r] + bias[n];
      }
}

// ---------------------------------------------------------------------------
// agg = shift_down( h2[:, :-1] @ We^T + be ); row l==0 -> 0.
// Writes agg f32 (for gated residual) and f16 (gate GEMM input).
// ---------------------------------------------------------------------------
__global__ __launch_bounds__(128) void k_gemm_agg(
    const _Float16* __restrict__ H2, const _Float16* __restrict__ Bt,
    const float* __restrict__ bias, float* __restrict__ agg32,
    _Float16* __restrict__ agg16) {
  const int K = 512;
  int lane = threadIdx.x & 31, wave = threadIdx.x >> 5;
  int mw = wave >> 1, nw = wave & 1;
  int m0 = blockIdx.x * 64 + mw * 32;
  int n0 = blockIdx.y * 128 + nw * 64;
  int r15 = lane & 15, hi = lane >> 4;
  v8f acc[2][4] = {};
  // shifted source rows: output row m reads h2 row m-1 (same batch); l==0 dummy
  int mr0 = m0 + r15;       int src0 = (mr0 & 2047) ? mr0 - 1 : mr0;
  int mr1 = m0 + 16 + r15;  int src1 = (mr1 & 2047) ? mr1 - 1 : mr1;
  gemm_tile(H2 + (long)src0 * K + hi * 8, H2 + (long)src1 * K + hi * 8,
            Bt + (n0 + r15) * K + hi * 8, K, K, acc);
#pragma unroll
  for (int i = 0; i < 2; ++i)
#pragma unroll
    for (int j = 0; j < 4; ++j)
#pragma unroll
      for (int r = 0; r < 8; ++r) {
        int m = m0 + i * 16 + hi * 8 + r;
        int n = n0 + j * 16 + r15;
        float v = ((m & 2047) != 0) ? (acc[i][j][r] + bias[n]) : 0.f;
        long idx = (long)m * 512 + n;
        agg32[idx] = v;
        agg16[idx] = (_Float16)v;
      }
}

// ---------------------------------------------------------------------------
// Gate + node update: g = sigmoid(h2 @ Wg_top^T + agg @ Wg_bot^T + bg)
// h2_new = h2 + g * agg.  WgT is [512,1024] (row n = [Wg[:512,n], Wg[512:,n]]).
// Writes h2 f32 (in place) and h2 f16 into ping-pong output buffer.
// ---------------------------------------------------------------------------
__global__ __launch_bounds__(128) void k_gemm_gate(
    const _Float16* __restrict__ H2in, const _Float16* __restrict__ Agg16,
    const _Float16* __restrict__ WgT, const float* __restrict__ bias,
    const float* __restrict__ agg32, float* __restrict__ h2f32,
    _Float16* __restrict__ H2out) {
  int lane = threadIdx.x & 31, wave = threadIdx.x >> 5;
  int mw = wave >> 1, nw = wave & 1;
  int m0 = blockIdx.x * 64 + mw * 32;
  int n0 = blockIdx.y * 128 + nw * 64;
  int r15 = lane & 15, hi = lane >> 4;
  v8f acc[2][4] = {};
  const _Float16* b0 = WgT + (n0 + r15) * 1024 + hi * 8;
  // pass 1: K = 0..511 over h2 (Wg top half)
  gemm_tile(H2in + (m0 + r15) * 512 + hi * 8,
            H2in + (m0 + 16 + r15) * 512 + hi * 8, b0, 512, 1024, acc);
  // pass 2: K = 512..1023 over agg (Wg bottom half)
  gemm_tile(Agg16 + (m0 + r15) * 512 + hi * 8,
            Agg16 + (m0 + 16 + r15) * 512 + hi * 8, b0 + 512, 512, 1024, acc);
#pragma unroll
  for (int i = 0; i < 2; ++i)
#pragma unroll
    for (int j = 0; j < 4; ++j)
#pragma unroll
      for (int r = 0; r < 8; ++r) {
        int m = m0 + i * 16 + hi * 8 + r;
        int n = n0 + j * 16 + r15;
        long idx = (long)m * 512 + n;
        float g = 1.f / (1.f + __expf(-(acc[i][j][r] + bias[n])));
        float hv = h2f32[idx] + g * agg32[idx];
        h2f32[idx] = hv;
        H2out[idx] = (_Float16)hv;
      }
}

// ---------------------------------------------------------------------------
// out = h + h2
// ---------------------------------------------------------------------------
__global__ void k_add(const float* __restrict__ a, const float* __restrict__ b,
                      float* __restrict__ out, int n) {
  int i = blockIdx.x * 256 + threadIdx.x;
  if (i < n) out[i] = a[i] + b[i];
}

// ---------------------------------------------------------------------------
extern "C" void kernel_launch(void* const* d_in, const int* in_sizes, int n_in,
                              void* d_out, int out_size, void* d_ws, size_t ws_size,
                              hipStream_t stream) {
  (void)in_sizes; (void)n_in; (void)out_size; (void)ws_size;
  const int M = 8 * 2048;         // 16384 rows
  const int D = 512;

  const float* x     = (const float*)d_in[0];
  /* d_in[1] = mask (causal, implemented analytically) */
  const float* ln1w  = (const float*)d_in[2];
  const float* ln1b  = (const float*)d_in[3];
  const float* ln2w  = (const float*)d_in[4];
  const float* ln2b  = (const float*)d_in[5];
  const float* Wqkv  = (const float*)d_in[6];
  const float* bqkv  = (const float*)d_in[7];
  const float* Wo    = (const float*)d_in[8];
  const float* bo    = (const float*)d_in[9];
  const float* We    = (const float*)d_in[10];
  const float* be    = (const float*)d_in[11];
  const float* Wg    = (const float*)d_in[12];
  const float* bg    = (const float*)d_in[13];
  float* out = (float*)d_out;

  char* ws = (char*)d_ws;
  size_t off = 0;
  auto alloc = [&](size_t bytes) { char* p = ws + off; off += (bytes + 255) & ~(size_t)255; return p; };
  const size_t F16 = (size_t)M * D * sizeof(_Float16);   // 16 MB
  const size_t F32 = (size_t)M * D * sizeof(float);      // 32 MB

  _Float16* hn    = (_Float16*)alloc(F16);
  _Float16* Qh    = (_Float16*)alloc(F16);
  _Float16* Kh    = (_Float16*)alloc(F16);
  _Float16* VT    = (_Float16*)alloc(F16);
  _Float16* ob    = (_Float16*)alloc(F16);
  _Float16* h2a   = (_Float16*)alloc(F16);
  _Float16* h2b   = (_Float16*)alloc(F16);
  _Float16* agg16 = (_Float16*)alloc(F16);
  float*    hbuf  = (float*)alloc(F32);
  float*    h2f32 = (float*)alloc(F32);
  float*    agg32 = (float*)alloc(F32);
  _Float16* WqkvT = (_Float16*)alloc((size_t)1536 * 512 * 2);
  _Float16* WoT   = (_Float16*)alloc((size_t)512 * 512 * 2);
  _Float16* WeT   = (_Float16*)alloc((size_t)512 * 512 * 2);
  _Float16* WgT   = (_Float16*)alloc((size_t)512 * 1024 * 2);

  // 1) weight transposes (f32 -> f16, W^T layout)
  k_transpose<<<(512 * 1536 + 255) / 256, 256, 0, stream>>>(Wqkv, WqkvT, 512, 1536);
  k_transpose<<<(512 * 512  + 255) / 256, 256, 0, stream>>>(Wo,   WoT,   512, 512);
  k_transpose<<<(512 * 512  + 255) / 256, 256, 0, stream>>>(We,   WeT,   512, 512);
  k_transpose<<<(1024 * 512 + 255) / 256, 256, 0, stream>>>(Wg,   WgT,   1024, 512);

  // 2) LN1 -> hn (f16)
  k_layernorm<<<M / 8, 256, 0, stream>>>(x, ln1w, ln1b, nullptr, hn);

  // 3) QKV GEMM, scatter to head-major Q/K and transposed V
  k_gemm_qkv<<<dim3(M / 64, 1536 / 128), 128, 0, stream>>>(hn, WqkvT, bqkv, Qh, Kh, VT);

  // 4) causal flash attention -> o (f16, [B,L,H*dh])
  k_attn<<<dim3(64, 2048 / 64), 128, 0, stream>>>(Qh, Kh, VT, ob);

  // 5) h = x + o @ Wo + bo
  k_gemm_out<<<dim3(M / 64, 512 / 128), 128, 0, stream>>>(ob, WoT, bo, x, hbuf);

  // 6) LN2 -> h2 (f32 + f16)
  k_layernorm<<<M / 8, 256, 0, stream>>>(hbuf, ln2w, ln2b, h2f32, h2a);

  // 7) TreeFFN x2 (ping-pong the f16 h2 buffer to avoid read/write races)
  _Float16* cur = h2a; _Float16* nxt = h2b;
  for (int it = 0; it < 2; ++it) {
    k_gemm_agg<<<dim3(M / 64, 512 / 128), 128, 0, stream>>>(cur, WeT, be, agg32, agg16);
    k_gemm_gate<<<dim3(M / 64, 512 / 128), 128, 0, stream>>>(cur, agg16, WgT, bg,
                                                             agg32, h2f32, nxt);
    _Float16* t = cur; cur = nxt; nxt = t;
  }

  // 8) out = h + h2
  k_add<<<(M * D + 255) / 256, 256, 0, stream>>>(hbuf, h2f32, out, M * D);
}
